// Decoder_62053687493166
// MI455X (gfx1250) — matrix operand order repaired and checked by reading, not compiled, not run on hardware
//
#include <hip/hip_runtime.h>
#include <math.h>

#define B_  32
#define T_  64
#define E_  256
#define H_  512
#define V_  32000
#define G_  (3*H_)    // 1536
#define TB_ (T_*B_)   // 2048

typedef _Float16 f16;
typedef _Float16 v4h  __attribute__((ext_vector_type(4)));
typedef _Float16 v8h  __attribute__((ext_vector_type(8)));
typedef _Float16 v16h __attribute__((ext_vector_type(16)));
typedef float    v4f  __attribute__((ext_vector_type(4)));
typedef float    v8f  __attribute__((ext_vector_type(8)));
typedef int      v4i  __attribute__((ext_vector_type(4)));

// ---------------------------------------------------------------------------
// gfx1250 async global->LDS copy support (GLOBAL_LOAD_ASYNC_TO_LDS_B128,
// tracked by ASYNCcnt). Signature (from compiler diagnostic):
//   void (v4i addrspace(1)*, v4i addrspace(3)*, imm int offset, imm int cpol)
// ---------------------------------------------------------------------------
#if defined(__has_builtin)
#if __has_builtin(__builtin_amdgcn_global_load_async_to_lds_b128)
#define HAVE_ASYNC_COPY 1
#endif
#endif
#ifndef HAVE_ASYNC_COPY
#define HAVE_ASYNC_COPY 0
#endif

#if HAVE_ASYNC_COPY
typedef __attribute__((address_space(1))) v4i as1_v4i;
typedef __attribute__((address_space(3))) v4i as3_v4i;
static __device__ inline void async_copy16(const f16* g, f16* l) {
    // per-lane: copy 16 bytes global -> LDS without VGPR round-trip
    __builtin_amdgcn_global_load_async_to_lds_b128((as1_v4i*)g, (as3_v4i*)l, 0, 0);
}
#if __has_builtin(__builtin_amdgcn_s_wait_asynccnt)
#define WAIT_ASYNC(n) __builtin_amdgcn_s_wait_asynccnt(n)
#else
#define WAIT_ASYNC(n) asm volatile("s_wait_asynccnt %0" :: "i"(n) : "memory")
#endif
#endif

static __device__ inline v16h cat8(v8h lo, v8h hi) {
    return __builtin_shufflevector(lo, hi, 0,1,2,3,4,5,6,7,8,9,10,11,12,13,14,15);
}

// A-fragment loader: 16x32 (MxK) f16 in contiguous k order (k = 16h + s), same as B.
// lanes 0-15:  M=lane,    K=0..15
// lanes 16-31: M=lane-16, K=16..31
// `A` points at element (m0, kc) of a row-major [*, lda] half array.
static __device__ inline v16h load_afrag(const f16* A, int lda) {
    const int lane = threadIdx.x & 31;
    const int m  = lane & 15;
    const int c0 = (lane < 16) ? 0 : 16;
    v8h lo = *(const v8h*)(A + m * lda + c0);
    v8h hi = *(const v8h*)(A + m * lda + c0 + 8);
    return cat8(lo, hi);
}

// B-fragment loader: 32x16 (KxN) f16. Contiguous k order, same as load_afrag:
// lanes 0-15: N=lane, K=0..15 ; lanes 16-31: N=lane-16, K=16..31.
// `Bp` points at element (n0, kc) of row-major [*, ldb] half array (W rows).
static __device__ inline v16h load_bfrag(const f16* Bp, int ldb) {
    const int lane = threadIdx.x & 31;
    const int n = lane & 15;
    const int c = (lane < 16) ? 0 : 16;
    return *(const v16h*)(Bp + (size_t)n * ldb + c);
}

static __device__ inline v8f wmma_f16(v16h a, v16h b, v8f c) {
    return __builtin_amdgcn_wmma_f32_16x16x32_f16(false, a, false, b, (short)0, c, false, false);
}

// ---------------------------------------------------------------------------
// Embedding gather + ReLU + fp16 convert.  Row = t*B + b.
// ---------------------------------------------------------------------------
__global__ __launch_bounds__(E_) void embed_relu_kernel(
    const float* __restrict__ emb, const int* __restrict__ tgt,
    f16* __restrict__ X)
{
    const int row = blockIdx.x;       // 0..2047
    const int t = row >> 5;           // row / 32
    const int b = row & 31;
    const int tok = (t == 0) ? 0 : tgt[b * T_ + (t - 1)];
    const float v = emb[(size_t)tok * E_ + threadIdx.x];
    X[(size_t)row * E_ + threadIdx.x] = (f16)fmaxf(v, 0.0f);
}

// ---------------------------------------------------------------------------
// fp32 -> fp16 conversion, 8 elements per thread (n must be multiple of 8).
// ---------------------------------------------------------------------------
__global__ __launch_bounds__(256) void cvt_f32_f16_kernel(
    const float* __restrict__ in, f16* __restrict__ out, size_t n)
{
    size_t i = ((size_t)blockIdx.x * 256 + threadIdx.x) * 8;
    if (i < n) {
        v4f a = *(const v4f*)(in + i);
        v4f b = *(const v4f*)(in + i + 4);
        v8h o;
        #pragma unroll
        for (int j = 0; j < 4; ++j) { o[j] = (f16)a[j]; o[4 + j] = (f16)b[j]; }
        *(v8h*)(out + i) = o;
    }
}

// ---------------------------------------------------------------------------
// WMMA GEMM:  C[M,N] = A[M,K](f16) * W[N,K](f16)^T + bias[N]   (fp32 out)
// block = 256 threads (8 waves). Block tile 64(M) x 128(N).
// Wave w owns N-tile w; computes 4 M-subtiles (B-fragment reused x4).
// A tile staged to LDS via async global->LDS copies, double buffered.
// grid = (N/128, M/64)
// ---------------------------------------------------------------------------
__global__ __launch_bounds__(256) void gemm_f16_wmma_kernel(
    const f16* __restrict__ A, const f16* __restrict__ W,
    const float* __restrict__ bias, float* __restrict__ C,
    int M, int N, int K)
{
#if HAVE_ASYNC_COPY
    __shared__ __align__(16) f16 As[2][64 * 32];
#else
    __shared__ __align__(16) f16 As[1][64 * 32];
#endif
    const int tid   = threadIdx.x;
    const int wave  = tid >> 5;
    const int lane  = tid & 31;
    const int mbase = blockIdx.y * 64;
    const int nbase = blockIdx.x * 128 + wave * 16;

    const int ldrow = tid >> 2;          // 0..63
    const int ldkg  = (tid & 3) * 8;     // 0,8,16,24
    const f16* Arow = A + (size_t)(mbase + ldrow) * K + ldkg;

    v8f acc0 = {}, acc1 = {}, acc2 = {}, acc3 = {};

#if HAVE_ASYNC_COPY
    const int nk = K >> 5;
    async_copy16(Arow, &As[0][ldrow * 32 + ldkg]);          // stage tile 0
    for (int i = 0; i < nk; ++i) {
        const int kc  = i << 5;
        const int cur = i & 1;
        if (i + 1 < nk) {
            async_copy16(Arow + kc + 32, &As[cur ^ 1][ldrow * 32 + ldkg]);
            WAIT_ASYNC(1);               // tile i done; tile i+1 still in flight
            __builtin_prefetch(W + (size_t)(nbase + (lane & 15)) * K + kc + 32, 0, 1);
        } else {
            WAIT_ASYNC(0);
        }
        __syncthreads();                 // all waves' copies for tile i visible

        v16h bf = load_bfrag(W + (size_t)nbase * K + kc, K);
        v16h a0 = load_afrag(&As[cur][ 0 * 32], 32);
        v16h a1 = load_afrag(&As[cur][16 * 32], 32);
        v16h a2 = load_afrag(&As[cur][32 * 32], 32);
        v16h a3 = load_afrag(&As[cur][48 * 32], 32);
        acc0 = wmma_f16(a0, bf, acc0);
        acc1 = wmma_f16(a1, bf, acc1);
        acc2 = wmma_f16(a2, bf, acc2);
        acc3 = wmma_f16(a3, bf, acc3);
        __syncthreads();                 // done reading before buffer reuse
    }
#else
    for (int kc = 0; kc < K; kc += 32) {
        *(v8h*)(&As[0][ldrow * 32 + ldkg]) = *(const v8h*)(Arow + kc);
        __syncthreads();
        if (kc + 32 < K)
            __builtin_prefetch(W + (size_t)(nbase + (lane & 15)) * K + kc + 32, 0, 1);
        v16h bf = load_bfrag(W + (size_t)nbase * K + kc, K);
        v16h a0 = load_afrag(&As[0][ 0 * 32], 32);
        v16h a1 = load_afrag(&As[0][16 * 32], 32);
        v16h a2 = load_afrag(&As[0][32 * 32], 32);
        v16h a3 = load_afrag(&As[0][48 * 32], 32);
        acc0 = wmma_f16(a0, bf, acc0);
        acc1 = wmma_f16(a1, bf, acc1);
        acc2 = wmma_f16(a2, bf, acc2);
        acc3 = wmma_f16(a3, bf, acc3);
        __syncthreads();
    }
#endif

    // Epilogue: C/D layout -> lane holds col = nbase + lane%16,
    // rows r (lanes 0-15) or 8+r (lanes 16-31) within each 16-row subtile.
    const int col = nbase + (lane & 15);
    const float bv = bias ? bias[col] : 0.0f;
    const int rbase = mbase + ((lane < 16) ? 0 : 8);
    v8f accs[4] = {acc0, acc1, acc2, acc3};
    #pragma unroll
    for (int mt = 0; mt < 4; ++mt) {
        #pragma unroll
        for (int r = 0; r < 8; ++r) {
            C[(size_t)(rbase + mt * 16 + r) * N + col] = accs[mt][r] + bv;
        }
    }
}

// ---------------------------------------------------------------------------
// Persistent GRU recurrence. 16 blocks x 384 threads (12 waves).
// Block `slice` owns h columns [slice*32, slice*32+32) and the matching
// r/z/n gate columns. Per step: gh = h[32,512] x W_hh_slice^T -> [32,96]
// via 12 16x16 WMMA tiles (2 M x 6 N), then gate math, then a device-wide
// atomic-counter barrier.
// ---------------------------------------------------------------------------
__global__ __launch_bounds__(384) void gru_kernel(
    const f16*   __restrict__ Whh,   // [1536][512] f16
    const float* __restrict__ GI,    // [(t*B+b)][1536], b_ih folded in
    const float* __restrict__ bhh,   // [1536]
    float*       __restrict__ hbuf,  // [2][B][H] ping-pong, hbuf[0] = h0
    f16*         __restrict__ Hseq,  // [(b*T+t)][H] f16
    float*       __restrict__ hTout, // d_out + B*T*V  (final hidden)
    unsigned int* __restrict__ cnt)  // zeroed before launch
{
    __shared__ __align__(16) f16 hs[B_ * H_];    // 32 KB
    __shared__ float ghs[B_ * 96];               // 12 KB

    const int tid  = threadIdx.x;
    const int wave = tid >> 5;
    const int lane = tid & 31;
    const int mt   = wave & 1;       // M tile: batch 0-15 / 16-31
    const int nt   = wave >> 1;      // 0..5 -> gate tiles
    const int slice = blockIdx.x;    // 0..15
    const int g    = nt >> 1;        // 0=r,1=z,2=n
    const int sub  = nt & 1;
    const int nrow0 = g * H_ + slice * 32 + sub * 16;  // W_hh row base

    for (int t = 0; t < T_; ++t) {
        const float* cur = hbuf + (size_t)(t & 1) * (B_ * H_);
        float*       nxt = hbuf + (size_t)((t + 1) & 1) * (B_ * H_);

        // Convert current h to f16 in LDS (cooperative).
        for (int i = tid; i < B_ * H_; i += 384) hs[i] = (f16)cur[i];
        __syncthreads();

        v8f acc = {};
        for (int kc = 0; kc < H_; kc += 32) {
            v16h af = load_afrag(hs + (mt * 16) * H_ + kc, H_);
            v16h bf = load_bfrag(Whh + (size_t)nrow0 * H_ + kc, H_);
            acc = wmma_f16(af, bf, acc);
        }

        // Scatter C tile into LDS gh buffer [b][96].
        {
            const int nl = lane & 15;
            const int r0 = (lane < 16) ? 0 : 8;
            #pragma unroll
            for (int r = 0; r < 8; ++r)
                ghs[(mt * 16 + r0 + r) * 96 + nt * 16 + nl] = acc[r];
        }
        __syncthreads();

        // Gate math for the 32x32 owned block.
        for (int idx = tid; idx < B_ * 32; idx += 384) {
            const int b = idx >> 5, j = idx & 31;
            const int col = slice * 32 + j;
            const float* gi = GI + (size_t)(t * B_ + b) * G_;
            const float ghr = ghs[b * 96 +       j] + bhh[          col];
            const float ghz = ghs[b * 96 + 32 +  j] + bhh[H_   +    col];
            const float ghn = ghs[b * 96 + 64 +  j] + bhh[2 * H_ +  col];
            const float rg = 1.0f / (1.0f + __expf(-(gi[          col] + ghr)));
            const float zg = 1.0f / (1.0f + __expf(-(gi[H_     +  col] + ghz)));
            const float ng = tanhf(gi[2 * H_ + col] + rg * ghn);
            const float hold = cur[b * H_ + col];
            const float hnew = (1.0f - zg) * ng + zg * hold;
            nxt[b * H_ + col] = hnew;
            Hseq[(size_t)(b * T_ + t) * H_ + col] = (f16)hnew;
            if (t == T_ - 1) hTout[b * H_ + col] = hnew;
        }

        // Device-wide barrier between timesteps.
        __threadfence();
        __syncthreads();
        if (tid == 0) {
            atomicAdd(cnt, 1u);
            const unsigned tgt = 16u * (unsigned)(t + 1);
            while (atomicAdd(cnt, 0u) < tgt) __builtin_amdgcn_s_sleep(2);
        }
        __syncthreads();
        __threadfence();
    }
}

// ---------------------------------------------------------------------------
// In-place row log_softmax over V=32000.  One block per (b,t) row.
// ---------------------------------------------------------------------------
__global__ __launch_bounds__(256) void logsoftmax_kernel(float* __restrict__ out)
{
    __shared__ float red[256];
    float* p = out + (size_t)blockIdx.x * V_;

    float mx = -3.402823466e38f;
    for (int i = threadIdx.x; i < V_; i += 256) mx = fmaxf(mx, p[i]);
    red[threadIdx.x] = mx; __syncthreads();
    for (int s = 128; s > 0; s >>= 1) {
        if (threadIdx.x < s) red[threadIdx.x] = fmaxf(red[threadIdx.x], red[threadIdx.x + s]);
        __syncthreads();
    }
    mx = red[0]; __syncthreads();

    float sum = 0.0f;
    for (int i = threadIdx.x; i < V_; i += 256) sum += __expf(p[i] - mx);
    red[threadIdx.x] = sum; __syncthreads();
    for (int s = 128; s > 0; s >>= 1) {
        if (threadIdx.x < s) red[threadIdx.x] += red[threadIdx.x + s];
        __syncthreads();
    }
    const float lse = mx + __logf(red[0]);
    __syncthreads();

    for (int i = threadIdx.x; i < V_; i += 256) p[i] = p[i] - lse;
}

// ---------------------------------------------------------------------------
// Workspace layout (bytes, 256-aligned)
// ---------------------------------------------------------------------------
#define OFS_X      ((size_t)0)                                  // 2048*256 f16
#define OFS_WIH16  (OFS_X     + (size_t)TB_ * E_ * 2)
#define OFS_WHH16  (OFS_WIH16 + (size_t)G_  * E_ * 2)
#define OFS_WOUT16 (OFS_WHH16 + (size_t)G_  * H_ * 2)
#define OFS_GI     (OFS_WOUT16+ (size_t)V_  * H_ * 2)
#define OFS_HSEQ   (OFS_GI    + (size_t)TB_ * G_ * 4)
#define OFS_HBUF   (OFS_HSEQ  + (size_t)TB_ * H_ * 2)
#define OFS_CNT    (OFS_HBUF  + (size_t)2 * B_ * H_ * 4)

extern "C" void kernel_launch(void* const* d_in, const int* in_sizes, int n_in,
                              void* d_out, int out_size, void* d_ws, size_t ws_size,
                              hipStream_t stream) {
    (void)in_sizes; (void)n_in; (void)out_size; (void)ws_size;

    // inputs in setup_inputs() order (encoder_outputs d_in[0] is unused by the math)
    const float* enc_hid   = (const float*)d_in[1];   // [1,B,H]
    const int*   target    = (const int*)  d_in[2];   // [B,T]
    const float* embedding = (const float*)d_in[3];   // [32000,256]
    const float* W_ih      = (const float*)d_in[4];   // [1536,256]
    const float* W_hh      = (const float*)d_in[5];   // [1536,512]
    const float* b_ih      = (const float*)d_in[6];
    const float* b_hh      = (const float*)d_in[7];
    const float* W_out     = (const float*)d_in[8];   // [32000,512]
    const float* b_out     = (const float*)d_in[9];

    float* out = (float*)d_out;                       // [B,T,V] ++ [1,B,H]
    char*  ws  = (char*)d_ws;

    f16*   Xf16   = (f16*)  (ws + OFS_X);
    f16*   Wih16  = (f16*)  (ws + OFS_WIH16);
    f16*   Whh16  = (f16*)  (ws + OFS_WHH16);
    f16*   Wout16 = (f16*)  (ws + OFS_WOUT16);
    float* GI     = (float*)(ws + OFS_GI);
    f16*   Hseq16 = (f16*)  (ws + OFS_HSEQ);
    float* hbuf   = (float*)(ws + OFS_HBUF);
    unsigned int* cnt = (unsigned int*)(ws + OFS_CNT);

    // Reset barrier counter (deterministic each call) and seed h0.
    (void)hipMemsetAsync(cnt, 0, sizeof(unsigned int), stream);
    (void)hipMemcpyAsync(hbuf, enc_hid, (size_t)B_ * H_ * sizeof(float),
                         hipMemcpyDeviceToDevice, stream);

    // 1) Embedding gather + ReLU -> f16 X, rows ordered (t*B + b)
    embed_relu_kernel<<<TB_, E_, 0, stream>>>(embedding, target, Xf16);

    // 2) Weight conversions to f16 (8 elems/thread)
    {
        size_t n;
        n = (size_t)G_ * E_;
        cvt_f32_f16_kernel<<<(unsigned)((n / 8 + 255) / 256), 256, 0, stream>>>(W_ih, Wih16, n);
        n = (size_t)G_ * H_;
        cvt_f32_f16_kernel<<<(unsigned)((n / 8 + 255) / 256), 256, 0, stream>>>(W_hh, Whh16, n);
        n = (size_t)V_ * H_;
        cvt_f32_f16_kernel<<<(unsigned)((n / 8 + 255) / 256), 256, 0, stream>>>(W_out, Wout16, n);
    }

    // 3) GI = X @ W_ih^T + b_ih   (M=2048, N=1536, K=256)
    gemm_f16_wmma_kernel<<<dim3(G_ / 128, TB_ / 64), 256, 0, stream>>>(
        Xf16, Wih16, b_ih, GI, TB_, G_, E_);

    // 4) GRU recurrence (persistent, 16 blocks, inter-step device barrier)
    gru_kernel<<<16, 384, 0, stream>>>(
        Whh16, GI, b_hh, hbuf, Hseq16, out + (size_t)B_ * T_ * V_, cnt);

    // 5) logits = Hseq @ W_out^T + b_out  -> straight into d_out [B,T,V]
    gemm_f16_wmma_kernel<<<dim3(V_ / 128, TB_ / 64), 256, 0, stream>>>(
        Hseq16, Wout16, b_out, out, TB_, V_, H_);

    // 6) in-place log_softmax per (b,t) row
    logsoftmax_kernel<<<TB_, 256, 0, stream>>>(out);
}
